// BiLSTM_CNN_CRF_52879637348465
// MI455X (gfx1250) — matrix operand order repaired and checked by reading, not compiled, not run on hardware
//
#include <hip/hip_runtime.h>
#include <hip/hip_bf16.h>
#include <math.h>

#define BATCH 32
#define SEQ   512
#define BL    (BATCH * SEQ)     // 16384 rows
#define EDIM  300
#define KPAD  320               // E padded to multiple of 32 (WMMA K)
#define HID   256
#define G4H   1024              // 4*H gate columns per direction
#define NTOT  2048              // both directions' gate columns
#define TAGS  4
#define WPITCH 272              // LDS row pitch (halves) to reduce bank conflicts
#define HSLOT (BATCH * HID)     // 8192 halves per timestep slot
#define HSZ   ((size_t)(SEQ + 1) * HSLOT)

typedef __attribute__((ext_vector_type(16))) _Float16 v16h;
typedef __attribute__((ext_vector_type(8)))  _Float16 v8h;
typedef __attribute__((ext_vector_type(8)))  float    v8f;

// Dense 16-bit A-matrix fragment (16x32 MxK): lane holds row m=lane&15,
// K chunks [k0+8g .. +7] and [k0+16+8g .. +7]  (g = lane>>4). Two b128 loads.
static __device__ __forceinline__ v16h frag_a(const _Float16* p) {
  v8h lo = *(const v8h*)p;
  v8h hi = *(const v8h*)(p + 16);
  return __builtin_shufflevector(lo, hi, 0,1,2,3,4,5,6,7,8,9,10,11,12,13,14,15);
}
// B-matrix fragment (32x16 KxN) loaded from row-major B^T (= weight[n][k]):
// lane holds column n=lane&15, K chunks [k0+8g .. +7] and [k0+16+8g .. +7].
static __device__ __forceinline__ v16h frag_b(const _Float16* p) {
  v8h lo = *(const v8h*)p;
  v8h hi = *(const v8h*)(p + 16);
  return __builtin_shufflevector(lo, hi, 0,1,2,3,4,5,6,7,8,9,10,11,12,13,14,15);
}

static __device__ __forceinline__ float sigf(float x) { return 1.0f / (1.0f + __expf(-x)); }

// ---------------- init: zero grid-barrier + h_seq step-0 slots ----------------
__global__ void init_k(_Float16* __restrict__ hseq, unsigned* __restrict__ bar) {
  int i = blockIdx.x * blockDim.x + threadIdx.x;
  if (i < HSLOT) {
    hseq[i]       = (_Float16)0.0f;
    hseq[HSZ + i] = (_Float16)0.0f;
  }
  if (i == 0) { bar[0] = 0u; bar[1] = 0u; }
}

// ---------------- embedding gather -> f16, K-padded ----------------
__global__ void embed_k(const int* __restrict__ ids, const float* __restrict__ emb,
                        _Float16* __restrict__ x) {
  long i = (long)blockIdx.x * blockDim.x + threadIdx.x;
  if (i >= (long)BL * KPAD) return;
  int row = (int)(i / KPAD), k = (int)(i % KPAD);
  float v = 0.0f;
  if (k < EDIM) v = emb[(size_t)ids[row] * EDIM + k];
  x[i] = (_Float16)v;
}

// ---------------- weight conversion f32 -> f16 ----------------
__global__ void cvt_wih_k(const float* __restrict__ wf, const float* __restrict__ wb,
                          _Float16* __restrict__ w) {
  int i = blockIdx.x * blockDim.x + threadIdx.x;
  if (i >= NTOT * KPAD) return;
  int n = i / KPAD, k = i % KPAD;
  float v = 0.0f;
  if (k < EDIM) v = (n < G4H) ? wf[n * EDIM + k] : wb[(n - G4H) * EDIM + k];
  w[i] = (_Float16)v;
}

__global__ void cvt_whh_k(const float* __restrict__ wf, const float* __restrict__ wb,
                          _Float16* __restrict__ w) {
  int i = blockIdx.x * blockDim.x + threadIdx.x;
  if (i >= 2 * G4H * HID) return;
  int d = i / (G4H * HID), r = i % (G4H * HID);
  w[i] = (_Float16)(d ? wb[r] : wf[r]);
}

// ---------------- input GEMM: xg = x @ Wih^T + b  (both dirs), WMMA f16 ----------------
// grid (BL/16, NTOT/512), block 256 (8 waves); wave computes 16 rows x 64 cols.
__global__ void __launch_bounds__(256) gemm_in_k(const _Float16* __restrict__ x,
                                                 const _Float16* __restrict__ w,
                                                 const float* __restrict__ bf,
                                                 const float* __restrict__ bb,
                                                 _Float16* __restrict__ xg) {
  const int wv = threadIdx.x >> 5, lane = threadIdx.x & 31;
  const int group = lane >> 4, ln = lane & 15;
  const int row0 = blockIdx.x * 16;
  const int col0 = blockIdx.y * 512 + wv * 64;

  v8f acc[4] = {};
  for (int k0 = 0; k0 < KPAD; k0 += 32) {
    v16h a = frag_a(x + (size_t)(row0 + ln) * KPAD + k0 + group * 8);
#pragma unroll
    for (int t = 0; t < 4; ++t) {
      v16h b = frag_b(w + (size_t)(col0 + t * 16 + ln) * KPAD + k0 + group * 8);
      acc[t] = __builtin_amdgcn_wmma_f32_16x16x32_f16(false, a, false, b,
                                                      (short)0, acc[t], false, false);
    }
  }
#pragma unroll
  for (int t = 0; t < 4; ++t) {
    int n = col0 + t * 16 + ln;
    int dir = n >> 10, cc = n & 1023;
    float bias = dir ? bb[cc] : bf[cc];
    _Float16* dst = xg + (size_t)dir * BL * G4H + cc;
#pragma unroll
    for (int v = 0; v < 8; ++v) {
      int m = row0 + group * 8 + v;
      dst[(size_t)m * G4H] = (_Float16)(acc[t][v] + bias);
    }
  }
}

// ---------------- persistent BiLSTM recurrence ----------------
// 16 blocks: dir = blk>>3, hidden group g = blk&7 (32 hidden units each).
// Block keeps Whh rows for its 128 gate columns in LDS; c-state in registers.
// Grid barrier (device-scope atomics) between timesteps.
__global__ void __launch_bounds__(128) lstm_k(const _Float16* __restrict__ xg,
                                              const _Float16* __restrict__ whh,
                                              _Float16* __restrict__ hseq,
                                              unsigned* __restrict__ bar) {
  extern __shared__ _Float16 sW[];  // 128 rows x WPITCH halves
  const int dir = blockIdx.x >> 3;
  const int g   = blockIdx.x & 7;
  const int tid = threadIdx.x;
  const int wv = tid >> 5, lane = tid & 31;
  const int group = lane >> 4, ln = lane & 15;
  const int mt = wv & 1;          // batch tile (rows mt*16..+15)
  const int st = (wv >> 1) & 1;   // hidden subtile (16 of 32 units)

  // Stage Whh slice: LDS row r = q*32+jj  <-  Whh[dir][q*256 + g*32 + jj][*]
  {
    const int q = tid >> 5, jj = tid & 31;
    const _Float16* src = whh + ((size_t)dir * G4H + q * 256 + g * 32 + jj) * HID;
#pragma unroll 4
    for (int k = 0; k < HID; k += 8)
      *(v8h*)&sW[tid * WPITCH + k] = *(const v8h*)&src[k];
  }
  __syncthreads();

  float c[8];
#pragma unroll
  for (int v = 0; v < 8; ++v) c[v] = 0.0f;

  const _Float16* xgd = xg + (size_t)dir * BL * G4H;
  _Float16* hs = hseq + (size_t)dir * HSZ;
  const int jj = st * 16 + ln;

  for (int s = 0; s < SEQ; ++s) {
    const int t = dir ? (SEQ - 1 - s) : s;
    const _Float16* hp = hs + (size_t)s * HSLOT + (size_t)(mt * 16 + ln) * HID;

    v8f acc[4] = {};
    for (int k0 = 0; k0 < HID; k0 += 32) {
      v16h a = frag_a(hp + k0 + group * 8);
#pragma unroll
      for (int q = 0; q < 4; ++q) {
        v16h b = frag_b(&sW[(q * 32 + st * 16 + ln) * WPITCH + k0 + group * 8]);
        acc[q] = __builtin_amdgcn_wmma_f32_16x16x32_f16(false, a, false, b,
                                                        (short)0, acc[q], false, false);
      }
    }

    _Float16* hout = hs + (size_t)(s + 1) * HSLOT;
#pragma unroll
    for (int v = 0; v < 8; ++v) {
      const int b = mt * 16 + group * 8 + v;  // batch row 0..31
      const _Float16* xr = xgd + ((size_t)b * SEQ + t) * G4H + g * 32 + jj;
      float gi = acc[0][v] + (float)xr[0];
      float gf = acc[1][v] + (float)xr[256];
      float gg = acc[2][v] + (float)xr[512];
      float go = acc[3][v] + (float)xr[768];
      float cn = sigf(gf) * c[v] + sigf(gi) * tanhf(gg);
      c[v] = cn;
      hout[(size_t)b * HID + g * 32 + jj] = (_Float16)(sigf(go) * tanhf(cn));
    }

    __threadfence();
    __syncthreads();
    if (tid == 0) {
      unsigned gen = __hip_atomic_load(&bar[1], __ATOMIC_ACQUIRE, __HIP_MEMORY_SCOPE_AGENT);
      if (__hip_atomic_fetch_add(&bar[0], 1u, __ATOMIC_ACQ_REL, __HIP_MEMORY_SCOPE_AGENT) == 15u) {
        __hip_atomic_store(&bar[0], 0u, __ATOMIC_RELAXED, __HIP_MEMORY_SCOPE_AGENT);
        __hip_atomic_fetch_add(&bar[1], 1u, __ATOMIC_RELEASE, __HIP_MEMORY_SCOPE_AGENT);
      } else {
        while (__hip_atomic_load(&bar[1], __ATOMIC_ACQUIRE, __HIP_MEMORY_SCOPE_AGENT) == gen)
          __builtin_amdgcn_s_sleep(8);
      }
    }
    __syncthreads();
  }
}

// ---------------- emissions: [h_f, h_b] @ W_out^T + b_out (T=4, scalar) ----------------
__global__ void emis_k(const _Float16* __restrict__ hseq, const float* __restrict__ Wout,
                       const float* __restrict__ bout, float* __restrict__ emis) {
  int i = blockIdx.x * blockDim.x + threadIdx.x;
  if (i >= BL) return;
  int b = i / SEQ, l = i % SEQ;
  const _Float16* hf = hseq + (size_t)(l + 1) * HSLOT + (size_t)b * HID;
  const _Float16* hb = hseq + HSZ + (size_t)(SEQ - l) * HSLOT + (size_t)b * HID;
  float a0 = bout[0], a1 = bout[1], a2 = bout[2], a3 = bout[3];
  for (int j = 0; j < HID; ++j) {
    float v = (float)hf[j];
    a0 += v * Wout[0 * 512 + j]; a1 += v * Wout[1 * 512 + j];
    a2 += v * Wout[2 * 512 + j]; a3 += v * Wout[3 * 512 + j];
  }
  for (int j = 0; j < HID; ++j) {
    float v = (float)hb[j];
    a0 += v * Wout[0 * 512 + 256 + j]; a1 += v * Wout[1 * 512 + 256 + j];
    a2 += v * Wout[2 * 512 + 256 + j]; a3 += v * Wout[3 * 512 + 256 + j];
  }
  float* e = emis + (size_t)i * TAGS;
  e[0] = a0; e[1] = a1; e[2] = a2; e[3] = a3;
}

// ---------------- Viterbi decode (one lane per batch element) ----------------
__global__ void viterbi_k(const float* __restrict__ emis, const int* __restrict__ mask,
                          const float* __restrict__ trans, const float* __restrict__ stt,
                          const float* __restrict__ ent, unsigned char* __restrict__ hist,
                          int* __restrict__ out) {
  int b = threadIdx.x;
  if (b >= BATCH) return;
  float tr[16];
#pragma unroll
  for (int i = 0; i < 16; ++i) tr[i] = trans[i];
  float sc[4];
#pragma unroll
  for (int j = 0; j < 4; ++j) sc[j] = stt[j] + emis[(size_t)b * SEQ * 4 + j];
  unsigned char* hb = hist + (size_t)b * (SEQ - 1) * 4;

  for (int t = 1; t < SEQ; ++t) {
    const float* e = emis + ((size_t)b * SEQ + t) * 4;
    int m = mask[b * SEQ + t];
    float ns[4];
#pragma unroll
    for (int j = 0; j < 4; ++j) {
      float best = sc[0] + tr[0 * 4 + j];
      int bi = 0;
#pragma unroll
      for (int i = 1; i < 4; ++i) {
        float v = sc[i] + tr[i * 4 + j];
        if (v > best) { best = v; bi = i; }   // strict > keeps first index (jnp.argmax)
      }
      ns[j] = best + e[j];
      hb[(t - 1) * 4 + j] = (unsigned char)bi;
    }
#pragma unroll
    for (int j = 0; j < 4; ++j) sc[j] = m ? ns[j] : sc[j];
  }
  int tag = 0;
  float bs = sc[0] + ent[0];
#pragma unroll
  for (int j = 1; j < 4; ++j) { float v = sc[j] + ent[j]; if (v > bs) { bs = v; tag = j; } }
  out[b * SEQ + SEQ - 1] = mask[b * SEQ + SEQ - 1] ? tag : 0;
  int cur = tag;
  for (int t = SEQ - 2; t >= 0; --t) {
    int prev = (int)hb[t * 4 + cur];
    if (mask[b * SEQ + t + 1]) cur = prev;
    out[b * SEQ + t] = mask[b * SEQ + t] ? cur : 0;
  }
}

extern "C" void kernel_launch(void* const* d_in, const int* in_sizes, int n_in,
                              void* d_out, int out_size, void* d_ws, size_t ws_size,
                              hipStream_t stream) {
  (void)in_sizes; (void)n_in; (void)out_size; (void)ws_size;
  const int*   word_ids = (const int*)  d_in[0];
  const int*   mask     = (const int*)  d_in[1];
  /* d_in[2] label_ids unused */
  const float* emb      = (const float*)d_in[3];
  const float* Wih_f    = (const float*)d_in[4];
  const float* Whh_f    = (const float*)d_in[5];
  const float* b_f      = (const float*)d_in[6];
  const float* Wih_b    = (const float*)d_in[7];
  const float* Whh_b    = (const float*)d_in[8];
  const float* b_b      = (const float*)d_in[9];
  const float* W_out    = (const float*)d_in[10];
  const float* b_out    = (const float*)d_in[11];
  const float* trans    = (const float*)d_in[12];
  const float* stt      = (const float*)d_in[13];
  const float* ent      = (const float*)d_in[14];

  char* ws = (char*)d_ws;
  unsigned* bar = (unsigned*)ws;
  size_t off = 256;
  _Float16* x16   = (_Float16*)(ws + off); off += (size_t)BL * KPAD * 2;        // 10.5 MB
  _Float16* wih16 = (_Float16*)(ws + off); off += (size_t)NTOT * KPAD * 2;      // 1.3 MB
  _Float16* whh16 = (_Float16*)(ws + off); off += (size_t)2 * G4H * HID * 2;    // 1.0 MB
  _Float16* xg    = (_Float16*)(ws + off); off += (size_t)2 * BL * G4H * 2;     // 67 MB
  _Float16* hseq  = (_Float16*)(ws + off); off += (size_t)2 * HSZ * 2;          // 16.8 MB
  float*    emis  = (float*)(ws + off);    off += (size_t)BL * TAGS * 4;        // 0.26 MB
  unsigned char* hist = (unsigned char*)(ws + off);                             // 64 KB

  init_k<<<(HSLOT + 255) / 256, 256, 0, stream>>>(hseq, bar);
  embed_k<<<(unsigned)(((size_t)BL * KPAD + 255) / 256), 256, 0, stream>>>(word_ids, emb, x16);
  cvt_wih_k<<<(NTOT * KPAD + 255) / 256, 256, 0, stream>>>(Wih_f, Wih_b, wih16);
  cvt_whh_k<<<(2 * G4H * HID + 255) / 256, 256, 0, stream>>>(Whh_f, Whh_b, whh16);
  gemm_in_k<<<dim3(BL / 16, NTOT / 512), 256, 0, stream>>>(x16, wih16, b_f, b_b, xg);
  lstm_k<<<16, 128, 128 * WPITCH * 2, stream>>>(xg, whh16, hseq, bar);
  emis_k<<<(BL + 127) / 128, 128, 0, stream>>>(hseq, W_out, b_out, emis);
  viterbi_k<<<1, 32, 0, stream>>>(emis, mask, trans, stt, ent, hist, (int*)d_out);
}
